// KNNGraphCrossAttention_9079560864224
// MI455X (gfx1250) — matrix operand order repaired and checked by reading, not compiled, not run on hardware
//
#include <hip/hip_runtime.h>

// ---------------------------------------------------------------------------
// KNN graph cross-attention for MI455X (gfx1250, wave32, WMMA).
//   BS=2, N=65536, C=256, KNN=16
// Kernel 1: fused projection GEMM (X @ W^T + b) via v_wmma_f32_16x16x32_bf16
// Kernel 2: per-query KNN attention + softmax + residual + layernorm (VALU),
//           with global_prefetch of V rows during the logit pass.
// ---------------------------------------------------------------------------

typedef __attribute__((ext_vector_type(16))) __bf16 v16bf;
typedef __attribute__((ext_vector_type(8)))  __bf16 v8bf;
typedef __attribute__((ext_vector_type(8)))  float  v8f;

#define C_DIM      256
#define N_PTS      65536
#define BS_DIM     2
#define KNN_K      16
#define LDS_STRIDE 264   // 256 + 8 bf16 pad: 528B rows -> conflict-free b128 LDS reads

// ---------------------------------------------------------------------------
// Projection GEMM:  Y[m, n] = sum_c X[m, c] * W[n, c] + bias[n]
//   M = BS*N = 131072, K = C = 256, N = C = 256
//   block: 256 threads = 8 waves, each wave computes a 16-row strip x 256 cols
//   grid.x = M / 128 = 1024
// ---------------------------------------------------------------------------
__global__ __launch_bounds__(256) void proj_gemm(
    const float* __restrict__ X,     // (M, 256) row-major fp32
    const float* __restrict__ W,     // (256, 256) row-major fp32 (out-ch major)
    const float* __restrict__ bias,  // (256,)
    float* __restrict__ Y)           // (M, 256) fp32
{
    __shared__ __bf16 sW[C_DIM * LDS_STRIDE];  // 132 KB of 320 KB WGP LDS

    const int tid = threadIdx.x;

    // Cooperative stage: W (256x256 fp32, row-major) -> bf16 LDS, padded rows.
    for (int e = tid; e < (C_DIM * C_DIM / 4); e += 256) {
        const int linear = e << 2;
        const int row = linear >> 8;
        const int col = linear & 255;
        const float4 w4 = ((const float4*)W)[e];
        __bf16* dst = &sW[row * LDS_STRIDE + col];
        dst[0] = (__bf16)w4.x;
        dst[1] = (__bf16)w4.y;
        dst[2] = (__bf16)w4.z;
        dst[3] = (__bf16)w4.w;
    }
    __syncthreads();

    const int wave = tid >> 5;
    const int lane = tid & 31;
    const int half = lane >> 4;   // 0: lanes 0-15, 1: lanes 16-31
    const int l16  = lane & 15;

    const int m0 = (blockIdx.x * 8 + wave) * 16;

    // ---- Load A strip (16 rows x 256 K) as 8 bf16 fragments (16x32 each) ----
    // ISA 16-bit A 16x32 layout: lanes 0-15 (M=l16) hold K = {k0..k0+7, k0+16..k0+23},
    // lanes 16-31 hold K = {k0+8..k0+15, k0+24..k0+31}.
    v16bf afrag[8];
    const float* xrow = X + (size_t)(m0 + l16) * C_DIM;
    for (int ks = 0; ks < 8; ++ks) {
        const int c1 = ks * 32 + half * 8;
        const int c2 = c1 + 16;
        const float4 f0 = *(const float4*)(xrow + c1);
        const float4 f1 = *(const float4*)(xrow + c1 + 4);
        const float4 f2 = *(const float4*)(xrow + c2);
        const float4 f3 = *(const float4*)(xrow + c2 + 4);
        v16bf a;
        a[0]  = (__bf16)f0.x; a[1]  = (__bf16)f0.y; a[2]  = (__bf16)f0.z; a[3]  = (__bf16)f0.w;
        a[4]  = (__bf16)f1.x; a[5]  = (__bf16)f1.y; a[6]  = (__bf16)f1.z; a[7]  = (__bf16)f1.w;
        a[8]  = (__bf16)f2.x; a[9]  = (__bf16)f2.y; a[10] = (__bf16)f2.z; a[11] = (__bf16)f2.w;
        a[12] = (__bf16)f3.x; a[13] = (__bf16)f3.y; a[14] = (__bf16)f3.z; a[15] = (__bf16)f3.w;
        afrag[ks] = a;
    }

    // ---- 16 column tiles x 8 K-steps of WMMA ----
    // B (32x16) layout: lane holds column l16; lanes 0-15 K = {k0..k0+7, k0+16..k0+23},
    // lanes 16-31 K = {k0+8..k0+15, k0+24..k0+31} (matches A).
    for (int ct = 0; ct < 16; ++ct) {
        const int n0 = ct * 16;
        v8f acc = {};
        const __bf16* wrow = &sW[(n0 + l16) * LDS_STRIDE];
        for (int ks = 0; ks < 8; ++ks) {
            const int c1 = ks * 32 + half * 8;
            union { v16bf v; v8bf h[2]; } bf;
            bf.h[0] = *(const v8bf*)(wrow + c1);      // 16B aligned (row 528B, c1*2 mult of 16B)
            bf.h[1] = *(const v8bf*)(wrow + c1 + 16);
            acc = __builtin_amdgcn_wmma_f32_16x16x32_bf16(
                false, afrag[ks], false, bf.v, (short)0, acc, false, false);
        }
        // C/D layout: lanes 0-15 -> M = v, N = l16; lanes 16-31 -> M = v + 8.
        const float bval = bias[n0 + l16];
        float* yb = Y + (size_t)(m0 + half * 8) * C_DIM + n0 + l16;
        #pragma unroll
        for (int v = 0; v < 8; ++v)
            yb[(size_t)v * C_DIM] = acc[v] + bval;
    }
}

// ---------------------------------------------------------------------------
// KNN attention + softmax + weighted-V + residual + layernorm.
//   One wave per query; lane holds 8 channels (c = lane*8 .. lane*8+7).
//   grid.x = BS*N/8 = 16384, block = 256 (8 waves).
//   V rows are prefetched (global_prefetch_b8) during the logit pass so the
//   post-softmax gather hits WGP$/L2.
// ---------------------------------------------------------------------------
__global__ __launch_bounds__(256) void knn_attn_ln(
    const float* __restrict__ Qp, const float* __restrict__ Kp,
    const float* __restrict__ Vp, const int* __restrict__ knn,
    const float* __restrict__ g,  const float* __restrict__ beta,
    float* __restrict__ out)
{
    const int lane = threadIdx.x & 31;
    const int wave = threadIdx.x >> 5;
    const int gq   = blockIdx.x * 8 + wave;     // 0 .. BS*N-1
    const int b    = gq >> 16;                  // N = 65536
    const int n    = gq & (N_PTS - 1);

    const size_t qoff = (size_t)gq * C_DIM;
    const int    c0   = lane * 8;

    float q[8];
    {
        const float4 q0 = *(const float4*)(Qp + qoff + c0);
        const float4 q1 = *(const float4*)(Qp + qoff + c0 + 4);
        q[0]=q0.x; q[1]=q0.y; q[2]=q0.z; q[3]=q0.w;
        q[4]=q1.x; q[5]=q1.y; q[6]=q1.z; q[7]=q1.w;
    }

    int myidx = 0;
    if (lane < KNN_K) myidx = knn[n * KNN_K + lane];

    const size_t bbase = (size_t)b * N_PTS * C_DIM;

    // Broadcast all 16 neighbor indices once; reused by both passes.
    int kidx[KNN_K];
    #pragma unroll
    for (int k = 0; k < KNN_K; ++k) kidx[k] = __shfl(myidx, k, 32);

    // ---- attention logits: dot(Qp[n], Kp[idx_k]) / sqrt(C) ----
    float attn[KNN_K];
    #pragma unroll
    for (int k = 0; k < KNN_K; ++k) {
        const float* krow = Kp + bbase + (size_t)kidx[k] * C_DIM + c0;
        // Warm WGP$/L2 for the matching V row (consumed after softmax).
        __builtin_prefetch((const void*)(Vp + bbase + (size_t)kidx[k] * C_DIM + c0), 0, 0);
        const float4 k0 = *(const float4*)krow;
        const float4 k1 = *(const float4*)(krow + 4);
        float s = q[0]*k0.x + q[1]*k0.y + q[2]*k0.z + q[3]*k0.w
                + q[4]*k1.x + q[5]*k1.y + q[6]*k1.z + q[7]*k1.w;
        #pragma unroll
        for (int m = 16; m >= 1; m >>= 1) s += __shfl_xor(s, m, 32);
        attn[k] = s * 0.0625f;                  // 1/sqrt(256)
    }

    // ---- softmax over 16 (replicated across lanes) ----
    float mx = attn[0];
    #pragma unroll
    for (int k = 1; k < KNN_K; ++k) mx = fmaxf(mx, attn[k]);
    float ssum = 0.f;
    #pragma unroll
    for (int k = 0; k < KNN_K; ++k) { attn[k] = __expf(attn[k] - mx); ssum += attn[k]; }
    const float inv = 1.0f / ssum;

    // ---- weighted sum of V rows ----
    float acc[8] = {0.f,0.f,0.f,0.f,0.f,0.f,0.f,0.f};
    #pragma unroll
    for (int k = 0; k < KNN_K; ++k) {
        const float w = attn[k] * inv;
        const float* vrow = Vp + bbase + (size_t)kidx[k] * C_DIM + c0;
        const float4 v0 = *(const float4*)vrow;
        const float4 v1 = *(const float4*)(vrow + 4);
        acc[0] += w*v0.x; acc[1] += w*v0.y; acc[2] += w*v0.z; acc[3] += w*v0.w;
        acc[4] += w*v1.x; acc[5] += w*v1.y; acc[6] += w*v1.z; acc[7] += w*v1.w;
    }

    // ---- residual + layernorm ----
    float r[8], psum = 0.f;
    #pragma unroll
    for (int i = 0; i < 8; ++i) { r[i] = acc[i] + q[i]; psum += r[i]; }
    #pragma unroll
    for (int m = 16; m >= 1; m >>= 1) psum += __shfl_xor(psum, m, 32);
    const float mu = psum * (1.0f / 256.0f);

    float pvar = 0.f;
    #pragma unroll
    for (int i = 0; i < 8; ++i) { const float d = r[i] - mu; pvar += d * d; }
    #pragma unroll
    for (int m = 16; m >= 1; m >>= 1) pvar += __shfl_xor(pvar, m, 32);
    const float rs = rsqrtf(pvar * (1.0f / 256.0f) + 1e-5f);

    const float4 g0 = *(const float4*)(g + c0);
    const float4 g1 = *(const float4*)(g + c0 + 4);
    const float4 b0 = *(const float4*)(beta + c0);
    const float4 b1 = *(const float4*)(beta + c0 + 4);

    float4 o0, o1;
    o0.x = (r[0]-mu)*rs*g0.x + b0.x;  o0.y = (r[1]-mu)*rs*g0.y + b0.y;
    o0.z = (r[2]-mu)*rs*g0.z + b0.z;  o0.w = (r[3]-mu)*rs*g0.w + b0.w;
    o1.x = (r[4]-mu)*rs*g1.x + b1.x;  o1.y = (r[5]-mu)*rs*g1.y + b1.y;
    o1.z = (r[6]-mu)*rs*g1.z + b1.z;  o1.w = (r[7]-mu)*rs*g1.w + b1.w;

    *(float4*)(out + qoff + c0)     = o0;
    *(float4*)(out + qoff + c0 + 4) = o1;
}

// ---------------------------------------------------------------------------
extern "C" void kernel_launch(void* const* d_in, const int* in_sizes, int n_in,
                              void* d_out, int out_size, void* d_ws, size_t ws_size,
                              hipStream_t stream) {
    const float* Q    = (const float*)d_in[0];
    const float* K    = (const float*)d_in[1];
    const float* V    = (const float*)d_in[2];
    const int*   knn  = (const int*)  d_in[3];
    const float* Wq   = (const float*)d_in[4];
    const float* bq   = (const float*)d_in[5];
    const float* Wk   = (const float*)d_in[6];
    const float* bk   = (const float*)d_in[7];
    const float* Wv   = (const float*)d_in[8];
    const float* bv   = (const float*)d_in[9];
    const float* ln_g = (const float*)d_in[10];
    const float* ln_b = (const float*)d_in[11];

    const size_t elems = (size_t)BS_DIM * N_PTS * C_DIM;   // 33,554,432
    float* Qp = (float*)d_ws;
    float* Kp = Qp + elems;
    float* Vp = Kp + elems;

    // M = 131072 rows; 8 waves * 16 rows = 128 rows/block -> 1024 blocks.
    proj_gemm<<<dim3(1024), dim3(256), 0, stream>>>(Q, Wq, bq, Qp);
    proj_gemm<<<dim3(1024), dim3(256), 0, stream>>>(K, Wk, bk, Kp);
    proj_gemm<<<dim3(1024), dim3(256), 0, stream>>>(V, Wv, bv, Vp);

    // BS*N = 131072 queries, 8 per block -> 16384 blocks.
    knn_attn_ln<<<dim3(16384), dim3(256), 0, stream>>>(
        Qp, Kp, Vp, knn, ln_g, ln_b, (float*)d_out);
}
